// SAModule_2482491097722
// MI455X (gfx1250) — matrix operand order repaired and checked by reading, not compiled, not run on hardware
//
#include <hip/hip_runtime.h>

typedef __attribute__((ext_vector_type(16))) _Float16 v16h;
typedef __attribute__((ext_vector_type(8)))  _Float16 v8h;
typedef __attribute__((ext_vector_type(8)))  float    v8f;
typedef __attribute__((ext_vector_type(4)))  float    v4f;

#define BB 8
#define CC 256
#define NN 4096
#define CQK 32

// ---- CDNA5 async global->LDS copy (ASYNCcnt) + raw wait helpers ----
__device__ __forceinline__ void async_b128(unsigned lds_off, const void* gbase, unsigned goff) {
  asm volatile("global_load_async_to_lds_b128 %0, %1, %2"
               :: "v"(lds_off), "v"(goff), "s"(gbase) : "memory");
}
__device__ __forceinline__ void wait_async0() {
  asm volatile("s_wait_asynccnt 0x0" ::: "memory");
}
__device__ __forceinline__ void wait_ds0() {
  asm volatile("s_wait_dscnt 0x0" ::: "memory");
}

// ---------------- Pack weights: Wh[320][256] f16 (+ bias[320] f32) ----------------
__global__ __launch_bounds__(256) void pack_w(
    const float* __restrict__ w1, const float* __restrict__ b1,
    const float* __restrict__ w2, const float* __restrict__ b2,
    const float* __restrict__ w3, const float* __restrict__ b3,
    _Float16* __restrict__ Wh, float* __restrict__ biasArr)
{
  const int idx = blockIdx.x * 256 + threadIdx.x;
  if (idx < 320 * 256) {
    const int m = idx >> 8, c = idx & 255;
    float v = (m < 32) ? w1[m * 256 + c]
            : (m < 64) ? w2[(m - 32) * 256 + c]
                       : w3[(m - 64) * 256 + c];
    Wh[idx] = (_Float16)v;
  }
  if (idx < 320) {
    biasArr[idx] = (idx < 32) ? b1[idx] : (idx < 64) ? b2[idx - 32] : b3[idx - 64];
  }
}

// ---------------- Transpose+convert: x[B][C][N] f32 -> Xh[B][N][C] f16 ----------------
__global__ __launch_bounds__(256) void x_to_f16t(
    const float* __restrict__ x, _Float16* __restrict__ Xh)
{
  __shared__ alignas(16) _Float16 t[64][72];     // 64x64 tile, padded
  const int bn = blockIdx.x;                     // b*64 + ngroup
  const int b  = bn >> 6;
  const int n0 = (bn & 63) << 6;
  const int c0 = blockIdx.y << 6;
  const int tid = threadIdx.x;
  const int nl = tid & 63;
  const int cl = tid >> 6;                       // 0..3
  const float* xp = x + ((size_t)b * CC + c0) * NN + n0;
#pragma unroll
  for (int i = 0; i < 16; i++) {
    const int c = cl + i * 4;
    t[nl][c] = (_Float16)xp[(size_t)c * NN + nl];  // coalesced reads along n
  }
  __syncthreads();
  const int nr = tid >> 2;                       // 0..63
  const int cc = (tid & 3) << 4;                 // 0,16,32,48
  _Float16* op = Xh + ((size_t)b * NN + n0 + nr) * CC + c0 + cc;
  const _Float16* src = &t[nr][cc];
  *(v8h*)(op)     = *(const v8h*)(src);
  *(v8h*)(op + 8) = *(const v8h*)(src + 8);
}

// ---------------- QKV projection as one WMMA GEMM ----------------
// Y[320 x 4096] = Wh[320 x 256] * X^T per batch; rows 0-31 -> Q, 32-63 -> K, 64-319 -> V.
// One wave = 1 mtile (16 out-ch) x 4 ntiles (64 cols); 8 K-steps x 4 = 32 WMMAs.
__global__ __launch_bounds__(256) void qkv_gemm(
    const _Float16* __restrict__ Xh, const _Float16* __restrict__ Wh,
    const float* __restrict__ biasArr,
    _Float16* __restrict__ Qh, _Float16* __restrict__ Kh, _Float16* __restrict__ Vh)
{
  const int tid = threadIdx.x;
  const int wave = tid >> 5, lane = tid & 31;
  const int h = lane >> 4, ln = lane & 15;
  const int task = blockIdx.x * 8 + wave;        // 8*20*64 = 10240 tasks
  const int b   = task / (20 * 64);
  const int rem = task % (20 * 64);
  const int mt  = rem / 64;                      // 0..19
  const int n0  = (rem % 64) * 64;

  const _Float16* wrow = Wh + (size_t)(mt * 16 + ln) * CC;
  const _Float16* xb   = Xh + (size_t)b * NN * CC;

  v8f acc[4];
#pragma unroll
  for (int nt = 0; nt < 4; nt++) { v8f z = {}; acc[nt] = z; }

#pragma unroll
  for (int k0 = 0; k0 < CC; k0 += 32) {
    v8h a0 = *(const v8h*)(wrow + k0 + h * 16);
    v8h a1 = *(const v8h*)(wrow + k0 + h * 16 + 8);
    v16h av;
#pragma unroll
    for (int i = 0; i < 8; i++) { av[i] = a0[i]; av[8 + i] = a1[i]; }
#pragma unroll
    for (int nt = 0; nt < 4; nt++) {
      v16h bx = *(const v16h*)(xb + (size_t)(n0 + nt * 16 + ln) * CC + k0 + h * 16);
      acc[nt] = __builtin_amdgcn_wmma_f32_16x16x32_f16(false, av, false, bx, (short)0, acc[nt], false, false);
    }
  }

  // bias for the 8 channels this lane holds (consecutive: mt*16 + 8h + r)
  v4f bv0 = *(const v4f*)(biasArr + mt * 16 + 8 * h);
  v4f bv1 = *(const v4f*)(biasArr + mt * 16 + 8 * h + 4);
  float bias[8] = { bv0[0], bv0[1], bv0[2], bv0[3], bv1[0], bv1[1], bv1[2], bv1[3] };

  if (mt < 4) {
    // Q/K: [B][N][32]; lane's 8 channels are consecutive -> one v8h store per ntile
    _Float16* dst = (mt < 2) ? Qh : Kh;
    const int cqb = (mt & 1) * 16 + 8 * h;
#pragma unroll
    for (int nt = 0; nt < 4; nt++) {
      const int n = n0 + nt * 16 + ln;
      v8h tv;
#pragma unroll
      for (int r = 0; r < 8; r++) tv[r] = (_Float16)(acc[nt][r] + bias[r]);
      *(v8h*)(dst + ((size_t)b * NN + n) * CQK + cqb) = tv;
    }
  } else {
    // V: [B][C][N]
    const int cb = (mt - 4) * 16 + 8 * h;
#pragma unroll
    for (int nt = 0; nt < 4; nt++) {
      const int n = n0 + nt * 16 + ln;
#pragma unroll
      for (int r = 0; r < 8; r++)
        Vh[((size_t)b * CC + cb + r) * NN + n] = (_Float16)(acc[nt][r] + bias[r]);
    }
  }
}

// ---------------- Fused flash attention + residual ----------------
// One wave = 16 query rows; block = 8 waves = 128 rows; grid = B*(N/128).
__global__ __launch_bounds__(256) void attn(
    const float* __restrict__ x, const _Float16* __restrict__ Qh,
    const _Float16* __restrict__ Kh, const _Float16* __restrict__ Vh,
    const float* __restrict__ gamma, float* __restrict__ out)
{
  constexpr int VP = 40;   // staged V panel: row pitch in halves (32 used + pad)
  constexpr int SP = 36;   // per-wave S tile: row pitch in floats (32 used + pad)
  __shared__ alignas(16) _Float16 vtile[2][CC * VP];  // 2 x 20 KiB, double buffered
  __shared__ alignas(16) float    stile[8][16 * SP];  // per-wave C->A relayout tile

  const int tid  = threadIdx.x;
  const int wave = tid >> 5;
  const int lane = tid & 31;
  const int h    = lane >> 4;
  const int ln   = lane & 15;
  const int b     = blockIdx.x >> 5;
  const int mbase = (blockIdx.x & 31) * 128 + wave * 16;

  const _Float16* kbase = Kh + (size_t)b * NN * CQK;
  const _Float16* vbase = Vh + (size_t)b * CC * NN;

  const _Float16* qrow = Qh + ((size_t)b * NN + (mbase + ln)) * CQK;
  v8h q0 = *(const v8h*)(qrow + h * 16);
  v8h q1 = *(const v8h*)(qrow + h * 16 + 8);
  v16h aq;
#pragma unroll
  for (int i = 0; i < 8; i++) { aq[i] = q0[i]; aq[8 + i] = q1[i]; }

  v16h ones;
#pragma unroll
  for (int i = 0; i < 16; i++) ones[i] = (_Float16)1.0f;

  float mi = -INFINITY;
  v8f li = {};
  v8f oacc[16];
#pragma unroll
  for (int ct = 0; ct < 16; ct++) { v8f z = {}; oacc[ct] = z; }

  float* srow_w = &stile[wave][0];
  const unsigned vrow_go = (unsigned)tid * (NN * 2);

  {
    unsigned lo = (unsigned)(size_t)&vtile[0][tid * VP];
#pragma unroll
    for (int k = 0; k < 4; k++) async_b128(lo + k * 16, vbase, vrow_go + k * 16);
    wait_async0();
    __syncthreads();
  }

  for (int n0 = 0; n0 < NN; n0 += 32) {
    const int buf = (n0 >> 5) & 1;
    const bool more = (n0 + 32) < NN;
    if (more) {
      unsigned lo = (unsigned)(size_t)&vtile[buf ^ 1][tid * VP];
      unsigned go = vrow_go + (unsigned)(n0 + 32) * 2;
#pragma unroll
      for (int k = 0; k < 4; k++) async_b128(lo + k * 16, vbase, go + k * 16);
    }

    v16h bk0 = *(const v16h*)(kbase + (size_t)(n0 + ln) * CQK + h * 16);
    v16h bk1 = *(const v16h*)(kbase + (size_t)(n0 + 16 + ln) * CQK + h * 16);
    __builtin_prefetch(kbase + (size_t)(n0 + 32 + ln) * CQK + h * 16, 0, 1);
    v8f z = {};
    v8f s0 = __builtin_amdgcn_wmma_f32_16x16x32_f16(false, aq, false, bk0, (short)0, z, false, false);
    v8f s1 = __builtin_amdgcn_wmma_f32_16x16x32_f16(false, aq, false, bk1, (short)0, z, false, false);

#pragma unroll
    for (int r = 0; r < 8; r++) {
      srow_w[(r + 8 * h) * SP + ln]      = s0[r];
      srow_w[(r + 8 * h) * SP + 16 + ln] = s1[r];
    }
    wait_ds0();

    const float* srd = &stile[wave][ln * SP];
    v4f sa0 = *(const v4f*)(srd + h * 8);
    v4f sa1 = *(const v4f*)(srd + h * 8 + 4);
    v4f sb0 = *(const v4f*)(srd + 16 + h * 8);
    v4f sb1 = *(const v4f*)(srd + 16 + h * 8 + 4);

    float t = fmaxf(fmaxf(fmaxf(sa0[0], sa0[1]), fmaxf(sa0[2], sa0[3])),
                    fmaxf(fmaxf(sa1[0], sa1[1]), fmaxf(sa1[2], sa1[3])));
    t = fmaxf(t, fmaxf(fmaxf(fmaxf(sb0[0], sb0[1]), fmaxf(sb0[2], sb0[3])),
                       fmaxf(fmaxf(sb1[0], sb1[1]), fmaxf(sb1[2], sb1[3]))));
    t = fmaxf(t, __shfl_xor(t, 16));
    const float mn = fmaxf(mi, t);
    const float alpha = __expf(mi - mn);
    mi = mn;

    v16h ap;
#pragma unroll
    for (int j = 0; j < 4; j++) {
      ap[j]      = (_Float16)__expf(sa0[j] - mn);
      ap[4 + j]  = (_Float16)__expf(sa1[j] - mn);
      ap[8 + j]  = (_Float16)__expf(sb0[j] - mn);
      ap[12 + j] = (_Float16)__expf(sb1[j] - mn);
    }

    if (__ballot(alpha != 1.0f)) {
      float alC[8];
#pragma unroll
      for (int r = 0; r < 8; r++) alC[r] = __shfl(alpha, r + 8 * h);
#pragma unroll
      for (int r = 0; r < 8; r++) li[r] *= alC[r];
#pragma unroll
      for (int ct = 0; ct < 16; ct++)
#pragma unroll
        for (int r = 0; r < 8; r++) oacc[ct][r] *= alC[r];
    }

    li = __builtin_amdgcn_wmma_f32_16x16x32_f16(false, ap, false, ones, (short)0, li, false, false);

    const _Float16* vt = &vtile[buf][0];
#pragma unroll
    for (int ct = 0; ct < 16; ct++) {
      const _Float16* vr = vt + (ct * 16 + ln) * VP + h * 8;
      v8h b0 = *(const v8h*)(vr);
      v8h b1 = *(const v8h*)(vr + 16);
      v16h bv;
#pragma unroll
      for (int i = 0; i < 8; i++) { bv[i] = b0[i]; bv[8 + i] = b1[i]; }
      oacc[ct] = __builtin_amdgcn_wmma_f32_16x16x32_f16(false, ap, false, bv, (short)0, oacc[ct], false, false);
    }

    if (more) {
      wait_async0();
      __syncthreads();
    }
  }

  const float g = gamma[0];
  float inv[8];
#pragma unroll
  for (int r = 0; r < 8; r++) inv[r] = 1.0f / li[r];
#pragma unroll
  for (int ct = 0; ct < 16; ct++) {
    const int c = ct * 16 + ln;
#pragma unroll
    for (int r = 0; r < 8; r++) {
      const int m = mbase + r + 8 * h;
      const size_t idx = ((size_t)b * CC + c) * NN + m;
      out[idx] = g * (oacc[ct][r] * inv[r]) + x[idx];
    }
  }
}

extern "C" void kernel_launch(void* const* d_in, const int* in_sizes, int n_in,
                              void* d_out, int out_size, void* d_ws, size_t ws_size,
                              hipStream_t stream) {
  const float* x  = (const float*)d_in[0];
  const float* w1 = (const float*)d_in[1];
  const float* b1 = (const float*)d_in[2];
  const float* w2 = (const float*)d_in[3];
  const float* b2 = (const float*)d_in[4];
  const float* w3 = (const float*)d_in[5];
  const float* b3 = (const float*)d_in[6];
  const float* gm = (const float*)d_in[7];
  float* out = (float*)d_out;

  _Float16* Qh = (_Float16*)d_ws;                       // B*N*32   (2 MiB)
  _Float16* Kh = Qh + (size_t)BB * NN * CQK;            // B*N*32   (2 MiB)
  _Float16* Vh = Kh + (size_t)BB * NN * CQK;            // B*C*N    (16 MiB)
  _Float16* Xh = Vh + (size_t)BB * CC * NN;             // B*N*C    (16 MiB)
  _Float16* Wh = Xh + (size_t)BB * NN * CC;             // 320*256  (160 KiB)
  float* biasA = (float*)(Wh + (size_t)320 * CC);       // 320 f32

  pack_w   <<<dim3(320),        dim3(256), 0, stream>>>(w1, b1, w2, b2, w3, b3, Wh, biasA);
  x_to_f16t<<<dim3(BB * 64, 4), dim3(256), 0, stream>>>(x, Xh);
  qkv_gemm <<<dim3(1280),       dim3(256), 0, stream>>>(Xh, Wh, biasA, Qh, Kh, Vh);
  attn     <<<dim3(BB * 32),    dim3(256), 0, stream>>>(x, Qh, Kh, Vh, gm, out);
}
